// MultiHeadSelfAttention_62173946577748
// MI455X (gfx1250) — compile-verified, matrix operand order checked by reading, not run on hardware
//
#include <hip/hip_runtime.h>
#include <hip/hip_bf16.h>

#define B_ 4
#define S_ 1024
#define D_ 2048
#define H_ 16
#define DH_ 128

typedef __attribute__((ext_vector_type(16))) __bf16 v16bf;
typedef __attribute__((ext_vector_type(8)))  float  v8f;

union FragAB { v16bf v; uint4 u[2]; __bf16 h[16]; };

__device__ __forceinline__ v8f wmma_bf16(v16bf a, v16bf b, v8f c) {
  return __builtin_amdgcn_wmma_f32_16x16x32_bf16(false, a, false, b, (short)0, c,
                                                 false, false);
}

// Wave-internal LDS ordering (waves in a block run different trip counts, so no
// block barrier allowed inside the attention key loop).
__device__ __forceinline__ void wave_lds_fence() {
  asm volatile("s_wait_dscnt 0" ::: "memory");
}

// CDNA5 async global->LDS DMA (VGLOBAL op 98): per-lane 16B transfer.
// vdst = LDS byte offset (low 32b of flat shared pointer), vaddr = global addr.
__device__ __forceinline__ void async_copy_b128(const void* gptr, void* lptr) {
  unsigned int loff = (unsigned int)(size_t)lptr;
  unsigned long long g = (unsigned long long)(size_t)gptr;
  asm volatile("global_load_async_to_lds_b128 %0, %1, off"
               :: "v"(loff), "v"(g) : "memory");
}
// Async loads complete in order -> waiting to <=5 drains the OLDER tile's five
// transfers while the just-issued five stay in flight (double buffering).
__device__ __forceinline__ void async_wait5() {
  asm volatile("s_wait_asynccnt 5" ::: "memory");
}
__device__ __forceinline__ void async_wait0() {
  asm volatile("s_wait_asynccnt 0" ::: "memory");
}

// ---------------------------------------------------------------- f32 -> bf16
__global__ void cvt_f32_to_bf16(const float* __restrict__ src,
                                __bf16* __restrict__ dst, int n) {
  int i = blockIdx.x * blockDim.x + threadIdx.x;
  int stride = gridDim.x * blockDim.x;
  for (; i < n; i += stride) dst[i] = (__bf16)src[i];
}

// ------------------------------------------------- f32 [K][N] -> bf16 [N][K]
__global__ void __launch_bounds__(256)
cvt_transpose_bf16(const float* __restrict__ src, __bf16* __restrict__ dst,
                   int K, int N) {
  __shared__ float tile[32][33];
  const int kb = blockIdx.y * 32, nb = blockIdx.x * 32;
  const int tx = threadIdx.x, ty = threadIdx.y;  // block (32, 8)
#pragma unroll
  for (int i = ty; i < 32; i += 8)
    tile[i][tx] = src[(size_t)(kb + i) * N + nb + tx];
  __syncthreads();
#pragma unroll
  for (int i = ty; i < 32; i += 8)
    dst[(size_t)(nb + i) * K + kb + tx] = (__bf16)tile[tx][i];
}

// ---------------------------------------------------------------- tiled GEMM
// C[M,N] = A[M,K] * B[K,N] with B supplied PRE-TRANSPOSED as Bt[N][K].
// bf16 inputs, f32 accumulate via WMMA. Block tile 64(M) x 256(N),
// 256 threads = 8 waves (2x4), wave tile 32x64, K step 32.
// Double-buffered async DMA staging: tile i+1 streams into the alternate LDS
// buffer while WMMA consumes tile i (s_wait_asynccnt 5 = drain older tile).
template <typename OutT>
__global__ void __launch_bounds__(256)
gemm_bf16(const __bf16* __restrict__ A, const __bf16* __restrict__ Bt,
          OutT* __restrict__ C, int M, int N, int K) {
  __shared__ alignas(16) __bf16 lds_a[2][64 * 32];   // row-major [64][32]
  __shared__ alignas(16) __bf16 lds_b[2][256 * 40];  // [n][k], stride 40 (80B)

  const int tid  = threadIdx.x;
  const int wave = tid >> 5, lane = tid & 31;
  const int wm = wave >> 2, wn = wave & 3;
  const int ln = lane & 15, hl = lane >> 4;
  const int blockN = blockIdx.x * 256;
  const int blockM = blockIdx.y * 64;
  const int arow = tid >> 2, agrp = (tid & 3) * 8;

  auto stage = [&](int k0, int buf) {
    // A tile 64x32: one async b128 per thread; B tile 256x32: four per thread.
    async_copy_b128(A + (size_t)(blockM + arow) * K + k0 + agrp,
                    &lds_a[buf][arow * 32 + agrp]);
#pragma unroll
    for (int it = 0; it < 4; it++) {
      int g   = tid + it * 256;
      int n   = g >> 2;          // 0..255
      int grp = (g & 3) * 8;     // 0,8,16,24
      async_copy_b128(Bt + (size_t)(blockN + n) * K + k0 + grp,
                      &lds_b[buf][n * 40 + grp]);
    }
  };

  v8f acc[2][4] = {};
  stage(0, 0);

  const int nk = K >> 5;
  for (int i = 0; i < nk; i++) {
    const int buf = i & 1;
    if (i + 1 < nk) {
      stage((i + 1) << 5, buf ^ 1);
      async_wait5();   // older tile (this buf) complete; next tile in flight
    } else {
      async_wait0();
    }
    __syncthreads();

    // ISA 16-bit A/B fragment layout: lane%16 = row/col, lane/16 selects
    // K runs [8h,8h+8) and [16+8h,16+8h+8)  -> two ds_load_b128 each.
    FragAB af[2], bf[4];
#pragma unroll
    for (int t = 0; t < 2; t++) {
      int row = wm * 32 + t * 16 + ln;
      af[t].u[0] = *(const uint4*)(&lds_a[buf][row * 32 + 8 * hl]);
      af[t].u[1] = *(const uint4*)(&lds_a[buf][row * 32 + 16 + 8 * hl]);
    }
#pragma unroll
    for (int t = 0; t < 4; t++) {
      int col = wn * 64 + t * 16 + ln;
      bf[t].u[0] = *(const uint4*)(&lds_b[buf][col * 40 + 8 * hl]);
      bf[t].u[1] = *(const uint4*)(&lds_b[buf][col * 40 + 16 + 8 * hl]);
    }
#pragma unroll
    for (int i2 = 0; i2 < 2; i2++)
#pragma unroll
      for (int j = 0; j < 4; j++)
        acc[i2][j] = wmma_bf16(af[i2].v, bf[j].v, acc[i2][j]);
    __syncthreads();  // all waves done reading buf before it is re-staged
  }

  // C/D layout: VGPR r -> row r + 8*(lane/16), col = lane%16.
#pragma unroll
  for (int i = 0; i < 2; i++)
#pragma unroll
    for (int j = 0; j < 4; j++)
#pragma unroll
      for (int r = 0; r < 8; r++) {
        int row = blockM + wm * 32 + i * 16 + r + 8 * hl;
        int col = blockN + wn * 64 + j * 16 + ln;
        C[(size_t)row * N + col] = (OutT)acc[i][j][r];
      }
}

// ------------------------------------------------------- RoPE + head split
// qkv bf16 [B,S,3D] -> Q,K (rotated) [B,H,S,DH]; V TRANSPOSED -> [B,H,DH,S]
// so attention's B-fragments become two b128 loads per 16x32 tile.
__global__ void rope_split(const __bf16* __restrict__ qkv,
                           const float* __restrict__ rope,
                           const int* __restrict__ pos,
                           __bf16* __restrict__ Q, __bf16* __restrict__ Kh,
                           __bf16* __restrict__ Vt) {
  int t = blockIdx.x * blockDim.x + threadIdx.x;
  if (t >= B_ * S_ * H_ * (DH_ / 2)) return;
  int p = t & 63; int r = t >> 6;
  int h = r & (H_ - 1); r >>= 4;
  int s = r & (S_ - 1); int b = r >> 10;
  int tp = pos[s];
  const float* m = rope + (((size_t)tp * 64 + p) * 4);  // [2][2] row-major
  size_t base = ((size_t)b * S_ + s) * (3 * D_) + h * DH_ + 2 * p;
  float q0 = (float)qkv[base],          q1 = (float)qkv[base + 1];
  float k0 = (float)qkv[base + D_],     k1 = (float)qkv[base + D_ + 1];
  float v0 = (float)qkv[base + 2 * D_], v1 = (float)qkv[base + 2 * D_ + 1];
  size_t bh = (size_t)b * H_ + h;
  size_t ob = (bh * S_ + s) * DH_ + 2 * p;
  Q[ob]      = (__bf16)(m[0] * q0 + m[1] * q1);
  Q[ob + 1]  = (__bf16)(m[2] * q0 + m[3] * q1);
  Kh[ob]     = (__bf16)(m[0] * k0 + m[1] * k1);
  Kh[ob + 1] = (__bf16)(m[2] * k0 + m[3] * k1);
  Vt[(bh * DH_ + 2 * p) * S_ + s]     = (__bf16)v0;
  Vt[(bh * DH_ + 2 * p + 1) * S_ + s] = (__bf16)v1;
}

// ---------------------------------------------------------- flash attention
// One wave per 16-row Q tile; 8 waves per block. Online softmax, 32-key chunks.
__global__ void __launch_bounds__(256)
flash_attn(const __bf16* __restrict__ Q, const __bf16* __restrict__ Kh,
           const __bf16* __restrict__ Vt, __bf16* __restrict__ ctx) {
  __shared__ alignas(16) float  s_sc[8][16 * 32];
  __shared__ alignas(16) __bf16 s_p[8][16 * 32];
  __shared__ float s_scale[8][16];
  __shared__ float s_l[8][16];

  const int tid  = threadIdx.x;
  const int wave = tid >> 5, lane = tid & 31;
  const int ln = lane & 15, hl = lane >> 4;
  int gw = blockIdx.x * 8 + wave;
  int qt = gw & 63;            // 64 q-tiles per (b,h)
  int bh = gw >> 6;            // b*H + h
  int qbase = qt * 16;
  const size_t head = (size_t)bh * S_ * DH_;

  // Q fragments for all 4 DH chunks (A layout, two b128 global loads each).
  FragAB qf[4];
  {
    const __bf16* qrow = Q + head + (size_t)(qbase + ln) * DH_;
#pragma unroll
    for (int c = 0; c < 4; c++) {
      qf[c].u[0] = *(const uint4*)(qrow + c * 32 + 8 * hl);
      qf[c].u[1] = *(const uint4*)(qrow + c * 32 + 16 + 8 * hl);
    }
  }

  v8f acc[8] = {};                 // 16x128 f32 context accumulator
  float m_row = -1e30f, l_row = 0.0f;
  const float sc = 0.08838834764831845f;  // 1/sqrt(128)

  const int nch = (qbase + 47) >> 5;      // causal chunk count
  for (int ch = 0; ch < nch; ch++) {
    int kbase = ch * 32;

    // S = Q K^T: per n-tile, keep 4 K fragments live (32 VGPRs) so the 8
    // loads batch into one clause and waits count down 4-deep.
    v8f st[2] = {};
#pragma unroll
    for (int tn = 0; tn < 2; tn++) {
      FragAB kf[4];
      const __bf16* krow = Kh + head + (size_t)(kbase + tn * 16 + ln) * DH_;
#pragma unroll
      for (int c = 0; c < 4; c++) {
        kf[c].u[0] = *(const uint4*)(krow + c * 32 + 8 * hl);
        kf[c].u[1] = *(const uint4*)(krow + c * 32 + 16 + 8 * hl);
      }
#pragma unroll
      for (int c = 0; c < 4; c++)
        st[tn] = wmma_bf16(qf[c].v, kf[c].v, st[tn]);
    }

    // Scale + causal mask, park score tile (16x32) in wave-private LDS.
#pragma unroll
    for (int r = 0; r < 8; r++) {
      int row = qbase + r + 8 * hl;
      int c0  = kbase + ln;
      float v0 = (c0 <= row)      ? st[0][r] * sc : -1e30f;
      float v1 = (c0 + 16 <= row) ? st[1][r] * sc : -1e30f;
      s_sc[wave][(r + 8 * hl) * 32 + ln]      = v0;
      s_sc[wave][(r + 8 * hl) * 32 + 16 + ln] = v1;
    }
    wave_lds_fence();

    // Online softmax: lane l < 16 owns row l.
    if (lane < 16) {
      float mx = m_row;
#pragma unroll
      for (int j = 0; j < 32; j++) mx = fmaxf(mx, s_sc[wave][lane * 32 + j]);
      float alpha = __expf(m_row - mx);
      float ls = 0.0f;
#pragma unroll
      for (int j = 0; j < 32; j++) {
        float p = __expf(s_sc[wave][lane * 32 + j] - mx);
        s_p[wave][lane * 32 + j] = (__bf16)p;
        ls += p;
      }
      l_row = alpha * l_row + ls;
      m_row = mx;
      s_scale[wave][lane] = alpha;
    }
    wave_lds_fence();

    float alr[8];
#pragma unroll
    for (int r = 0; r < 8; r++) alr[r] = s_scale[wave][r + 8 * hl];

    // P as A fragment (16x32 bf16) from LDS.
    FragAB pf;
    pf.u[0] = *(const uint4*)(&s_p[wave][ln * 32 + 8 * hl]);
    pf.u[1] = *(const uint4*)(&s_p[wave][ln * 32 + 16 + 8 * hl]);

    // V fragments (dh-major -> two contiguous b128 loads each); load all 8
    // first, then rescale+accumulate.
    FragAB vf[8];
#pragma unroll
    for (int t = 0; t < 8; t++) {
      const __bf16* vcol = Vt + ((size_t)bh * DH_ + t * 16 + ln) * S_ + kbase;
      vf[t].u[0] = *(const uint4*)(vcol + 8 * hl);
      vf[t].u[1] = *(const uint4*)(vcol + 16 + 8 * hl);
    }
#pragma unroll
    for (int t = 0; t < 8; t++) {
#pragma unroll
      for (int r = 0; r < 8; r++) acc[t][r] *= alr[r];
      acc[t] = wmma_bf16(pf.v, vf[t].v, acc[t]);
    }
  }

  if (lane < 16) s_l[wave][lane] = l_row;
  wave_lds_fence();
  float rinv[8];
#pragma unroll
  for (int r = 0; r < 8; r++) rinv[r] = 1.0f / s_l[wave][r + 8 * hl];

  int h = bh & (H_ - 1);
  int b = bh >> 4;
#pragma unroll
  for (int t = 0; t < 8; t++)
#pragma unroll
    for (int r = 0; r < 8; r++) {
      int row = qbase + r + 8 * hl;
      size_t off = ((size_t)b * S_ + row) * D_ + h * DH_ + t * 16 + ln;
      ctx[off] = (__bf16)(acc[t][r] * rinv[r]);
    }
}

// --------------------------------------------------------------------- host
extern "C" void kernel_launch(void* const* d_in, const int* in_sizes, int n_in,
                              void* d_out, int out_size, void* d_ws,
                              size_t ws_size, hipStream_t stream) {
  (void)in_sizes; (void)n_in; (void)out_size; (void)ws_size;
  const float* x     = (const float*)d_in[0];
  const float* w_qkv = (const float*)d_in[1];
  const float* w_out = (const float*)d_in[2];
  const float* rope  = (const float*)d_in[3];
  const int*   pos   = (const int*)d_in[4];
  float* out = (float*)d_out;

  char* ws = (char*)d_ws;
  size_t off = 0;
  auto alloc = [&](size_t bytes) {
    void* p = ws + off;
    off = (off + bytes + 255) & ~(size_t)255;
    return p;
  };
  __bf16* xb    = (__bf16*)alloc((size_t)B_ * S_ * D_ * 2);
  __bf16* wqkvt = (__bf16*)alloc((size_t)D_ * 3 * D_ * 2);   // [3D][D] bf16
  __bf16* woutt = (__bf16*)alloc((size_t)D_ * D_ * 2);       // [D][D]  bf16
  __bf16* qkvb  = (__bf16*)alloc((size_t)B_ * S_ * 3 * D_ * 2);
  __bf16* Qh    = (__bf16*)alloc((size_t)B_ * H_ * S_ * DH_ * 2);
  __bf16* Kh2   = (__bf16*)alloc((size_t)B_ * H_ * S_ * DH_ * 2);
  __bf16* Vt    = (__bf16*)alloc((size_t)B_ * H_ * DH_ * S_ * 2);
  __bf16* ctxb  = (__bf16*)alloc((size_t)B_ * S_ * D_ * 2);

  // 1) down-convert x; transpose+convert weights to [N][K] bf16 (paid once).
  cvt_f32_to_bf16<<<2048, 256, 0, stream>>>(x, xb, B_ * S_ * D_);
  cvt_transpose_bf16<<<dim3((3 * D_) / 32, D_ / 32), dim3(32, 8), 0, stream>>>(
      w_qkv, wqkvt, D_, 3 * D_);
  cvt_transpose_bf16<<<dim3(D_ / 32, D_ / 32), dim3(32, 8), 0, stream>>>(
      w_out, woutt, D_, D_);

  // 2) qkv = x @ w_qkv : [4096,2048] x [2048,6144] -> bf16
  gemm_bf16<__bf16><<<dim3((3 * D_) / 256, (B_ * S_) / 64), 256, 0, stream>>>(
      xb, wqkvt, qkvb, B_ * S_, 3 * D_, D_);

  // 3) RoPE + split into heads (V transposed to dh-major)
  rope_split<<<(B_ * S_ * H_ * (DH_ / 2)) / 256, 256, 0, stream>>>(
      qkvb, rope, pos, Qh, Kh2, Vt);

  // 4) fused causal flash attention -> ctx bf16 [B,S,D]
  flash_attn<<<(B_ * H_ * (S_ / 16)) / 8, 256, 0, stream>>>(Qh, Kh2, Vt, ctxb);

  // 5) out = ctx @ w_out : [4096,2048] x [2048,2048] -> f32
  gemm_bf16<float><<<dim3(D_ / 256, (B_ * S_) / 64), 256, 0, stream>>>(
      ctxb, woutt, out, B_ * S_, D_, D_);
}
